// PPG_9594956939771
// MI455X (gfx1250) — matrix operand order repaired and checked by reading, compiled, not run on hardware
//
#include <hip/hip_runtime.h>

typedef __attribute__((ext_vector_type(16))) _Float16 v16h;
typedef __attribute__((ext_vector_type(8)))  _Float16 v8h;
typedef __attribute__((ext_vector_type(8)))  float    v8f;

#define IMG 256
#define HW  65536
#define PW  258          // padded spatial width (1-pixel halo each side)
#define PHW 66564        // 258*258

// ---------------- Gaussian blur matrix (replicate boundary), sigma = 256, R = 768 ----------------

__global__ void build_g_kernel(float* __restrict__ g) {
  __shared__ float red[256];
  int tid = threadIdx.x;
  float s = 0.f;
  for (int i = tid; i < 1537; i += 256) {
    float off = (float)(i - 768);
    float v = expf(-off * off * (1.0f / (2.0f * 256.0f * 256.0f)));
    g[i] = v; s += v;
  }
  red[tid] = s; __syncthreads();
  for (int st = 128; st > 0; st >>= 1) { if (tid < st) red[tid] += red[tid + st]; __syncthreads(); }
  float S = red[0];
  for (int i = tid; i < 1537; i += 256) g[i] = g[i] / S;
}

__global__ void build_K_kernel(const float* __restrict__ g, float* __restrict__ K) {
  int i = blockIdx.x, j = threadIdx.x;
  float v;
  if (j == 0) {
    float s = 0.f;
    for (int off = -768; off <= -i; ++off) s += g[off + 768];
    v = s;
  } else if (j == IMG - 1) {
    float s = 0.f;
    for (int off = (IMG - 1) - i; off <= 768; ++off) s += g[off + 768];
    v = s;
  } else {
    v = g[j - i + 768];
  }
  K[i * IMG + j] = v;
}

__global__ void quantize_kernel(const float* __restrict__ x, float* __restrict__ q, int n) {
  int i = blockIdx.x * blockDim.x + threadIdx.x;
  if (i < n) q[i] = floorf(fminf(fmaxf(x[i] * 255.f, 0.f), 255.f)) * (1.f / 255.f);
}

// t1[h,w] = sum_k K[h,k] * q[k,w]
__global__ void blurH_kernel(const float* __restrict__ K, const float* __restrict__ q,
                             float* __restrict__ t1) {
  int bc = blockIdx.x >> 8; int h = blockIdx.x & 255; int w = threadIdx.x;
  const float* qb = q + (size_t)bc * HW;
  const float* Kr = K + h * IMG;
  float s = 0.f;
  for (int k = 0; k < IMG; ++k) s = fmaf(Kr[k], qb[k * IMG + w], s);
  t1[(size_t)bc * HW + h * IMG + w] = s;
}

// A[h,w] = round(clip(sum_k K[w,k] * t1[h,k],0,1)*255)/255
__global__ void blurW_kernel(const float* __restrict__ K, const float* __restrict__ t1,
                             float* __restrict__ A) {
  int bc = blockIdx.x >> 8; int h = blockIdx.x & 255; int w = threadIdx.x;
  const float* tb = t1 + (size_t)bc * HW + h * IMG;
  const float* Kr = K + w * IMG;
  float s = 0.f;
  for (int k = 0; k < IMG; ++k) s = fmaf(Kr[k], tb[k], s);
  s = rintf(fminf(fmaxf(s, 0.f), 1.f) * 255.f) * (1.f / 255.f);
  A[(size_t)bc * HW + h * IMG + w] = s;
}

// Zero the 1-pixel halo of a padded NHWC buffer at channel stride ICp
__global__ void zero_border_kernel(_Float16* __restrict__ buf, int ICp) {
  int i = blockIdx.x * blockDim.x + threadIdx.x;
  const int total = 8 * 1028;   // border pixels per image: 2*258 + 2*256
  if (i >= total) return;
  int b = i / 1028; int p = i % 1028;
  int py, px;
  if (p < 258)      { py = 0;           px = p;       }
  else if (p < 516) { py = PW - 1;      px = p - 258; }
  else if (p < 772) { py = p - 516 + 1; px = 0;       }
  else              { py = p - 772 + 1; px = PW - 1;  }
  _Float16* dst = buf + ((size_t)(b * PW + py) * PW + px) * ICp;
  v8h z;
  #pragma unroll
  for (int r = 0; r < 8; ++r) z[r] = (_Float16)0.f;
  for (int c = 0; c < ICp; c += 8) *(v8h*)(dst + c) = z;
}

// NCHW f32 -> padded NHWC f16 (IC padded to 32): inA = A + x, inT = x
__global__ void finalize_kernel(const float* __restrict__ x, const float* __restrict__ A,
                                _Float16* __restrict__ inA, _Float16* __restrict__ inT, int n) {
  int i = blockIdx.x * blockDim.x + threadIdx.x;
  if (i >= n) return;
  int b = i >> 16; int p = i & 65535;
  int y = p >> 8;  int xx = p & 255;
  size_t pix = ((size_t)(b * PW + y + 1) * PW + xx + 1) * 32;
  _Float16* pa = inA + pix;
  _Float16* pt = inT + pix;
  for (int c = 0; c < 3; ++c) {
    size_t idx = (size_t)(b * 3 + c) * HW + p;
    float xv = x[idx];
    pa[c] = (_Float16)(A[idx] + xv);
    pt[c] = (_Float16)xv;
  }
  for (int c = 3; c < 32; ++c) { pa[c] = (_Float16)0.f; pt[c] = (_Float16)0.f; }
}

// ---------------- Weight packing into per-lane WMMA A-fragment layout (16x32 f16) ----------------
// packed index = (((tap*OCT + t)*KT + kt)*32 + lane)*16 + e
__global__ void pack_w_kernel(const float* __restrict__ w, _Float16* __restrict__ p,
                              int OC, int IC, int OCT, int KT) {
  int idx = blockIdx.x * blockDim.x + threadIdx.x;
  int total = 9 * OCT * KT * 32 * 16;
  if (idx >= total) return;
  int e = idx & 15; int lane = (idx >> 4) & 31; int rest = idx >> 9;
  int kt = rest % KT; rest /= KT; int t = rest % OCT; int tap = rest / OCT;
  int half = lane >> 4; int v = e >> 1; int pos = e & 1;
  // contiguous A layout (matches B fragment): K = 16*half + e
  int k = half * 16 + 2 * v + pos;
  int ic = kt * 32 + k;
  int oc = t * 16 + (lane & 15);
  int ky = tap / 3, kx = tap % 3;
  float val = 0.f;
  if (oc < OC && ic < IC) val = w[((oc * IC + ic) * 3 + ky) * 3 + kx];
  p[idx] = (_Float16)val;
}

// ---------------- 3x3 SAME conv as implicit GEMM via v_wmma_f32_16x16x32_f16 ----------------
// Input/output activations are spatially padded (258x258) NHWC f16; halo is pre-zeroed,
// so every fragment load is unconditional: no divergence, no zero-fill in the hot loop.
// One wave: 16 output pixels (row segment) x OCT*16 output channels, K = KT*32 input channels.
template<int OCT, int KT, bool RELU, bool F32OUT>
__global__ void conv3x3_wmma_kernel(const _Float16* __restrict__ in,
                                    const _Float16* __restrict__ wpack,
                                    const float* __restrict__ bias,
                                    _Float16* __restrict__ out16,
                                    float* __restrict__ outf32,
                                    int OCreal) {
  const int ICp = KT * 32;
  const int OCp = OCT * 16;
  int tid  = blockIdx.x * blockDim.x + threadIdx.x;
  int wave = tid >> 5;
  int lane = tid & 31;
  int xt = wave & 15;
  int y  = (wave >> 4) & 255;
  int b  = wave >> 12;
  int x0 = xt << 4;
  int n    = lane & 15;
  int half = lane >> 4;

  v8f acc[OCT];
  #pragma unroll
  for (int t = 0; t < OCT; ++t)
    #pragma unroll
    for (int r = 0; r < 8; ++r) acc[t][r] = 0.f;

  for (int tap = 0; tap < 9; ++tap) {
    int py = y + tap / 3;        // padded row:  (y+1) + dy,  dy = tap/3 - 1
    int px = x0 + n + tap % 3;   // padded col:  (x+1) + dx,  dx = tap%3 - 1
    const _Float16* brow = in + ((size_t)(b * PW + py) * PW + px) * ICp + half * 16;
    #pragma unroll
    for (int kt = 0; kt < KT; ++kt) {
      v16h bfrag = *(const v16h*)(brow + kt * 32);
      const _Float16* abase = wpack + (size_t)((tap * OCT) * KT + kt) * 512 + lane * 16;
      #pragma unroll
      for (int t = 0; t < OCT; ++t) {
        v16h afrag = *(const v16h*)(abase + (size_t)t * KT * 512);
        acc[t] = __builtin_amdgcn_wmma_f32_16x16x32_f16(
            /*neg_a=*/false, afrag, /*neg_b=*/false, bfrag,
            /*c_mod=*/(short)0, acc[t], /*reuse_a=*/false, /*reuse_b=*/false);
      }
    }
  }

  if (F32OUT) {
    // NCHW f32 final output, only OCreal channels valid
    #pragma unroll
    for (int r = 0; r < 8; ++r) {
      int m = half * 8 + r;
      if (m < OCreal) {
        float vv = acc[0][r] + bias[m];
        outf32[(((size_t)b * OCreal + m) << 16) + y * 256 + x0 + n] = vv;
      }
    }
  } else {
    size_t obase = ((size_t)(b * PW + y + 1) * PW + x0 + n + 1) * OCp + half * 8;
    #pragma unroll
    for (int t = 0; t < OCT; ++t) {
      v8h ov;
      #pragma unroll
      for (int r = 0; r < 8; ++r) {
        float vv = acc[t][r] + bias[t * 16 + half * 8 + r];
        if (RELU) vv = fmaxf(vv, 0.f);
        ov[r] = (_Float16)vv;
      }
      *(v8h*)(out16 + obase + t * 16) = ov;
    }
  }
}

// ---------------- Orchestration ----------------

extern "C" void kernel_launch(void* const* d_in, const int* in_sizes, int n_in,
                              void* d_out, int out_size, void* d_ws, size_t ws_size,
                              hipStream_t stream) {
  (void)in_sizes; (void)n_in; (void)out_size; (void)ws_size;
  const float* x    = (const float*)d_in[0];
  const float* a_w1 = (const float*)d_in[1];  const float* a_b1 = (const float*)d_in[2];
  const float* a_w2 = (const float*)d_in[3];  const float* a_b2 = (const float*)d_in[4];
  const float* a_w3 = (const float*)d_in[5];  const float* a_b3 = (const float*)d_in[6];
  const float* a_wf = (const float*)d_in[7];  const float* a_bf = (const float*)d_in[8];
  const float* t_w1 = (const float*)d_in[9];  const float* t_b1 = (const float*)d_in[10];
  const float* t_w2 = (const float*)d_in[11]; const float* t_b2 = (const float*)d_in[12];
  const float* t_w3 = (const float*)d_in[13]; const float* t_b3 = (const float*)d_in[14];
  const float* t_wf = (const float*)d_in[15]; const float* t_bf = (const float*)d_in[16];

  char* ws = (char*)d_ws;
  size_t off = 0;
  auto take = [&](size_t bytes) -> char* {
    off = (off + 255) & ~(size_t)255;
    char* r = ws + off;
    off += bytes;
    return r;
  };

  float*    g    = (float*)take(1537 * 4);
  float*    Kb   = (float*)take((size_t)IMG * IMG * 4);
  float*    q    = (float*)take(24ull * HW * 4);
  float*    t1b  = (float*)take(24ull * HW * 4);
  float*    Ab   = (float*)take(24ull * HW * 4);
  _Float16* inA  = (_Float16*)take(8ull * PHW * 32 * 2);
  _Float16* inT  = (_Float16*)take(8ull * PHW * 32 * 2);
  _Float16* buf0 = (_Float16*)take(8ull * PHW * 128 * 2);
  _Float16* buf1 = (_Float16*)take(8ull * PHW * 128 * 2);
  _Float16* wp   = (_Float16*)take(512 * 1024);

  // get_A pipeline
  build_g_kernel<<<1, 256, 0, stream>>>(g);
  build_K_kernel<<<256, 256, 0, stream>>>(g, Kb);
  quantize_kernel<<<(24 * HW + 255) / 256, 256, 0, stream>>>(x, q, 24 * HW);
  blurH_kernel<<<24 * 256, 256, 0, stream>>>(Kb, q, t1b);
  blurW_kernel<<<24 * 256, 256, 0, stream>>>(Kb, t1b, Ab);
  finalize_kernel<<<(8 * HW + 255) / 256, 256, 0, stream>>>(x, Ab, inA, inT, 8 * HW);

  const int convBlocks = 4096;   // 8*256*16 waves * 32 lanes / 256 threads
  const int bordBlocks = (8 * 1028 + 255) / 256;
  float* out_a = (float*)d_out;
  float* out_t = (float*)d_out + 8ull * 3 * HW;

  // aNet: (A+x) -> 64 -> 128 -> 128 -> 3
  zero_border_kernel<<<bordBlocks, 256, 0, stream>>>(inA, 32);
  pack_w_kernel<<<(9 * 4 * 1 * 512 + 255) / 256, 256, 0, stream>>>(a_w1, wp, 64, 3, 4, 1);
  conv3x3_wmma_kernel<4, 1, true,  false><<<convBlocks, 256, 0, stream>>>(inA,  wp, a_b1, buf0, nullptr, 64);
  zero_border_kernel<<<bordBlocks, 256, 0, stream>>>(buf0, 64);
  pack_w_kernel<<<(9 * 8 * 2 * 512 + 255) / 256, 256, 0, stream>>>(a_w2, wp, 128, 64, 8, 2);
  conv3x3_wmma_kernel<8, 2, true,  false><<<convBlocks, 256, 0, stream>>>(buf0, wp, a_b2, buf1, nullptr, 128);
  zero_border_kernel<<<bordBlocks, 256, 0, stream>>>(buf1, 128);
  pack_w_kernel<<<(9 * 8 * 4 * 512 + 255) / 256, 256, 0, stream>>>(a_w3, wp, 128, 128, 8, 4);
  conv3x3_wmma_kernel<8, 4, true,  false><<<convBlocks, 256, 0, stream>>>(buf1, wp, a_b3, buf0, nullptr, 128);
  zero_border_kernel<<<bordBlocks, 256, 0, stream>>>(buf0, 128);
  pack_w_kernel<<<(9 * 1 * 4 * 512 + 255) / 256, 256, 0, stream>>>(a_wf, wp, 3, 128, 1, 4);
  conv3x3_wmma_kernel<1, 4, false, true ><<<convBlocks, 256, 0, stream>>>(buf0, wp, a_bf, nullptr, out_a, 3);

  // tNet: x -> 64 -> 128 -> 128 -> 1
  zero_border_kernel<<<bordBlocks, 256, 0, stream>>>(inT, 32);
  pack_w_kernel<<<(9 * 4 * 1 * 512 + 255) / 256, 256, 0, stream>>>(t_w1, wp, 64, 3, 4, 1);
  conv3x3_wmma_kernel<4, 1, true,  false><<<convBlocks, 256, 0, stream>>>(inT,  wp, t_b1, buf0, nullptr, 64);
  zero_border_kernel<<<bordBlocks, 256, 0, stream>>>(buf0, 64);
  pack_w_kernel<<<(9 * 8 * 2 * 512 + 255) / 256, 256, 0, stream>>>(t_w2, wp, 128, 64, 8, 2);
  conv3x3_wmma_kernel<8, 2, true,  false><<<convBlocks, 256, 0, stream>>>(buf0, wp, t_b2, buf1, nullptr, 128);
  zero_border_kernel<<<bordBlocks, 256, 0, stream>>>(buf1, 128);
  pack_w_kernel<<<(9 * 8 * 4 * 512 + 255) / 256, 256, 0, stream>>>(t_w3, wp, 128, 128, 8, 4);
  conv3x3_wmma_kernel<8, 4, true,  false><<<convBlocks, 256, 0, stream>>>(buf1, wp, t_b3, buf0, nullptr, 128);
  zero_border_kernel<<<bordBlocks, 256, 0, stream>>>(buf0, 128);
  pack_w_kernel<<<(9 * 1 * 4 * 512 + 255) / 256, 256, 0, stream>>>(t_wf, wp, 1, 128, 1, 4);
  conv3x3_wmma_kernel<1, 4, false, true ><<<convBlocks, 256, 0, stream>>>(buf0, wp, t_bf, nullptr, out_t, 1);
}
